// ConditionalLM_4612794876170
// MI455X (gfx1250) — matrix operand order repaired and checked by reading, not compiled, not run on hardware
//
#include <hip/hip_runtime.h>

typedef __bf16 bf16_t;
typedef __attribute__((ext_vector_type(16))) __bf16 v16bf;
typedef __attribute__((ext_vector_type(8)))  __bf16 v8bf;
typedef __attribute__((ext_vector_type(8)))  float  v8f;
typedef int v4i __attribute__((vector_size(16)));

#define BATCH 256
#define NH    512
#define NC    1024
#define VOCAB 32002
#define G3    1536
#define TLEN  15
#define KSTEPS (NH / 32)     // 16 k-slabs of 32 for the logits GEMM
#define LROW  40             // padded LDS row stride in bf16 elems (80 B)

// --------------------------------------------------------------------------
// async global->LDS copy (gfx1250 GLOBAL_LOAD_ASYNC_TO_LDS_B128, ASYNCcnt)
// --------------------------------------------------------------------------
#if defined(__has_builtin)
#if __has_builtin(__builtin_amdgcn_global_load_async_to_lds_b128) && \
    __has_builtin(__builtin_amdgcn_s_wait_asynccnt)
#define HAS_ASYNC_LDS 1
#endif
#endif
#ifndef HAS_ASYNC_LDS
#define HAS_ASYNC_LDS 0
#endif

__device__ __forceinline__ void cp16_async(void* lds, const void* g) {
#if HAS_ASYNC_LDS
  __builtin_amdgcn_global_load_async_to_lds_b128(
      (__attribute__((address_space(1))) v4i*)g,
      (__attribute__((address_space(3))) v4i*)lds,
      0, 0);
#else
  *(int4*)lds = *(const int4*)g;     // synchronous fallback (visibility via barrier)
#endif
}
#if HAS_ASYNC_LDS
#define ASYNC_WAIT(n) __builtin_amdgcn_s_wait_asynccnt(n)
#else
#define ASYNC_WAIT(n)
#endif

// ---------------------------------------------------------------------------
// 64x64 output tile per wave, bf16 WMMA, weights row-major [N, K] (X @ W.T),
// operands streamed straight from L2 (used by the small GEMMs)
// ---------------------------------------------------------------------------
__device__ __forceinline__ void wmma_gemm_64x64(
    const bf16_t* __restrict__ A, int mbase, int lda,
    const bf16_t* __restrict__ W, int nbase, int nmax, int ldw,
    int K, v8f acc[4][4])
{
  const int lane = threadIdx.x & 31;
  const int half = lane >> 4;
  const int l16  = lane & 15;

#pragma unroll
  for (int m = 0; m < 4; m++)
#pragma unroll
    for (int n = 0; n < 4; n++)
#pragma unroll
      for (int i = 0; i < 8; i++) acc[m][n][i] = 0.0f;

  for (int kb = 0; kb < K; kb += 32) {
    v16bf bfrag[4];
#pragma unroll
    for (int nsub = 0; nsub < 4; nsub++) {
      int nrow = nbase + nsub * 16 + l16;
      if (nrow >= nmax) nrow = nmax - 1;
      bfrag[nsub] = *(const v16bf*)(W + (size_t)nrow * ldw + kb + half * 16);
    }
#pragma unroll
    for (int msub = 0; msub < 4; msub++) {
      const bf16_t* ap = A + (size_t)(mbase + msub * 16 + l16) * lda + kb + half * 16;
      v8bf lo = *(const v8bf*)(ap);
      v8bf hi = *(const v8bf*)(ap + 8);
      v16bf afrag;
#pragma unroll
      for (int i = 0; i < 8; i++) { afrag[i] = lo[i]; afrag[i + 8] = hi[i]; }
#pragma unroll
      for (int nsub = 0; nsub < 4; nsub++)
        acc[msub][nsub] = __builtin_amdgcn_wmma_f32_16x16x32_bf16(
            false, afrag, false, bfrag[nsub], (short)0, acc[msub][nsub], false, false);
    }
  }
}

// --------------------------------------------------------------------------
__global__ void k_cvt_bf16(const float* __restrict__ src, bf16_t* __restrict__ dst, int n) {
  int i = blockIdx.x * blockDim.x + threadIdx.x;
  if (i < n) dst[i] = (bf16_t)src[i];
}

__global__ void k_seed(const int* __restrict__ caption, int* __restrict__ preds) {
  int b = blockIdx.x * blockDim.x + threadIdx.x;
  if (b < BATCH) preds[b * TLEN] = caption[b];
}

// ---------------------------------------------------------------------------
// h0 = img @ W_cond.T + b_cond   (M=256, N=512, K=1024)
// ---------------------------------------------------------------------------
__global__ __launch_bounds__(128, 1)
void k_h0(const bf16_t* __restrict__ imgb, const bf16_t* __restrict__ wcondb,
          const float* __restrict__ b_cond, float* __restrict__ h, bf16_t* __restrict__ hb) {
  const int mbase = (threadIdx.x >> 5) * 64;
  const int nbase = blockIdx.x * 64;
  v8f acc[4][4];
  wmma_gemm_64x64(imgb, mbase, NC, wcondb, nbase, NH, NC, NC, acc);
  const int lane = threadIdx.x & 31, half = lane >> 4, l16 = lane & 15;
#pragma unroll
  for (int msub = 0; msub < 4; msub++)
#pragma unroll
    for (int nsub = 0; nsub < 4; nsub++)
#pragma unroll
      for (int r = 0; r < 8; r++) {
        int row = mbase + msub * 16 + r + half * 8;
        int col = nbase + nsub * 16 + l16;
        float v = acc[msub][nsub][r] + b_cond[col];
        h[row * NH + col]  = v;
        hb[row * NH + col] = (bf16_t)v;
      }
}

// ---------------------------------------------------------------------------
__global__ void k_gather(const int* __restrict__ preds, int t,
                         const float* __restrict__ emb, bf16_t* __restrict__ xb) {
  int idx = blockIdx.x * blockDim.x + threadIdx.x;
  int b = idx >> 9, j = idx & 511;
  int tok = preds[b * TLEN + (t - 1)];
  xb[idx] = (bf16_t)emb[(size_t)tok * NH + j];
}

// ---------------------------------------------------------------------------
// gi = x @ w_ih.T + b_ih ; gh = h @ w_hh.T + b_hh   (blockIdx.y picks GEMM)
// ---------------------------------------------------------------------------
__global__ __launch_bounds__(128, 1)
void k_gates(const bf16_t* __restrict__ xb, const bf16_t* __restrict__ hb,
             const bf16_t* __restrict__ wihb, const bf16_t* __restrict__ whhb,
             const float* __restrict__ b_ih, const float* __restrict__ b_hh,
             float* __restrict__ gi, float* __restrict__ gh) {
  const bf16_t* A; const bf16_t* W; const float* bias; float* out;
  if (blockIdx.y == 0) { A = xb; W = wihb; bias = b_ih; out = gi; }
  else                 { A = hb; W = whhb; bias = b_hh; out = gh; }
  const int mbase = (threadIdx.x >> 5) * 64;
  const int nbase = blockIdx.x * 64;
  v8f acc[4][4];
  wmma_gemm_64x64(A, mbase, NH, W, nbase, G3, NH, NH, acc);
  const int lane = threadIdx.x & 31, half = lane >> 4, l16 = lane & 15;
#pragma unroll
  for (int msub = 0; msub < 4; msub++)
#pragma unroll
    for (int nsub = 0; nsub < 4; nsub++)
#pragma unroll
      for (int r = 0; r < 8; r++) {
        int row = mbase + msub * 16 + r + half * 8;
        int col = nbase + nsub * 16 + l16;
        out[row * G3 + col] = acc[msub][nsub][r] + bias[col];
      }
}

// ---------------------------------------------------------------------------
__global__ void k_combine(const float* __restrict__ gi, const float* __restrict__ gh,
                          float* __restrict__ h, bf16_t* __restrict__ hb,
                          unsigned long long* __restrict__ amax) {
  int idx = blockIdx.x * blockDim.x + threadIdx.x;
  if (idx < BATCH) amax[idx] = 0ull;
  int b = idx >> 9, j = idx & 511;
  float ir = gi[b * G3 + j],         hr = gh[b * G3 + j];
  float iz = gi[b * G3 + 512 + j],   hz = gh[b * G3 + 512 + j];
  float in_ = gi[b * G3 + 1024 + j], hn = gh[b * G3 + 1024 + j];
  float r = 1.0f / (1.0f + __expf(-(ir + hr)));
  float z = 1.0f / (1.0f + __expf(-(iz + hz)));
  float n = tanhf(in_ + r * hn);
  float hv = (1.0f - z) * n + z * h[b * NH + j];
  h[b * NH + j]  = hv;
  hb[b * NH + j] = (bf16_t)hv;
}

__device__ __forceinline__ unsigned int fkey(float v) {
  unsigned int b = __float_as_uint(v);
  return (b & 0x80000000u) ? ~b : (b | 0x80000000u);
}

// ---------------------------------------------------------------------------
// logits = h_new @ W_pred.T + b_pred ; argmax via packed atomicMax.
// 256-thread block computes a 256(M) x 128(N) tile; 32-wide K-slabs of A and B
// are staged into LDS with async DMA, double-buffered; 8 waves consume:
//   wave w: M rows [ (w&3)*64, +64 ), N cols [ (w>>2)*64, +64 ) of the block.
// ---------------------------------------------------------------------------
__global__ __launch_bounds__(256, 1)
void k_logits(const bf16_t* __restrict__ hb, const bf16_t* __restrict__ wpb,
              const float* __restrict__ b_pred, unsigned long long* __restrict__ amax) {
  __shared__ bf16_t Ash[2][256 * LROW];   // 40 KB: 256 rows x 32 K (padded)
  __shared__ bf16_t Bsh[2][128 * LROW];   // 20 KB: 128 rows x 32 K (padded)

  const int tid  = threadIdx.x;
  const int wave = tid >> 5;
  const int lane = tid & 31;
  const int half = lane >> 4, l16 = lane & 15;
  const int mbase = (wave & 3) * 64;       // batch rows (global == LDS-local)
  const int nloc  = (wave >> 2) * 64;      // LDS-local vocab rows
  const int nbase = blockIdx.x * 128;      // global vocab base of this block

  // cooperative slab copy: A row = tid (4 x b128), B row = tid>>1 (2 x b128)
  const int brow_l = tid >> 1;
  int brow_g = nbase + brow_l; if (brow_g >= VOCAB) brow_g = VOCAB - 1;
  const char* gA = (const char*)(hb + (size_t)tid * NH);
  const char* gB = (const char*)(wpb + (size_t)brow_g * NH);
  const int   bc0 = (tid & 1) * 2;

  v8f acc[4][4];
#pragma unroll
  for (int m = 0; m < 4; m++)
#pragma unroll
    for (int n = 0; n < 4; n++)
#pragma unroll
      for (int i = 0; i < 8; i++) acc[m][n][i] = 0.0f;

  auto stage_copy = [&](int st, int kb) {
    char* la = (char*)&Ash[st][tid * LROW];
    const char* ga = gA + kb * 2;
#pragma unroll
    for (int c = 0; c < 4; c++) cp16_async(la + c * 16, ga + c * 16);
    char* lb = (char*)&Bsh[st][brow_l * LROW];
    const char* gb = gB + kb * 2;
#pragma unroll
    for (int c = 0; c < 2; c++) cp16_async(lb + (bc0 + c) * 16, gb + (bc0 + c) * 16);
  };

  stage_copy(0, 0);
  for (int ks = 0; ks < KSTEPS; ks++) {
    const int st = ks & 1;
    if (ks + 1 < KSTEPS) {          // prefetch next slab, then wait for current
      stage_copy(st ^ 1, (ks + 1) * 32);
      ASYNC_WAIT(6);                // 6 copies of next slab still in flight
    } else {
      ASYNC_WAIT(0);
    }
    __syncthreads();

    v16bf bfrag[4];
#pragma unroll
    for (int nsub = 0; nsub < 4; nsub++) {
      const bf16_t* bp = &Bsh[st][(nloc + nsub * 16 + l16) * LROW + half * 16];
      v8bf b0 = *(const v8bf*)(bp);
      v8bf b1 = *(const v8bf*)(bp + 8);
#pragma unroll
      for (int i = 0; i < 8; i++) { bfrag[nsub][i] = b0[i]; bfrag[nsub][i + 8] = b1[i]; }
    }
#pragma unroll
    for (int msub = 0; msub < 4; msub++) {
      const bf16_t* ap = &Ash[st][(mbase + msub * 16 + l16) * LROW + half * 16];
      v8bf lo = *(const v8bf*)(ap);
      v8bf hi = *(const v8bf*)(ap + 8);
      v16bf afrag;
#pragma unroll
      for (int i = 0; i < 8; i++) { afrag[i] = lo[i]; afrag[i + 8] = hi[i]; }
#pragma unroll
      for (int nsub = 0; nsub < 4; nsub++)
        acc[msub][nsub] = __builtin_amdgcn_wmma_f32_16x16x32_bf16(
            false, afrag, false, bfrag[nsub], (short)0, acc[msub][nsub], false, false);
    }
    __syncthreads();
  }

  // bias + row-wise argmax (packed key, lowest index wins ties)
#pragma unroll
  for (int msub = 0; msub < 4; msub++)
#pragma unroll
    for (int r = 0; r < 8; r++) {
      float bestv = -3.4e38f;
      int   bestc = 0;
#pragma unroll
      for (int nsub = 0; nsub < 4; nsub++) {
        int col = nbase + nloc + nsub * 16 + l16;
        float v = (col < VOCAB) ? (acc[msub][nsub][r] + b_pred[col]) : -3.4e38f;
        if (v > bestv || (v == bestv && col < bestc)) { bestv = v; bestc = col; }
      }
#pragma unroll
      for (int m = 1; m < 16; m <<= 1) {
        float ov = __shfl_xor(bestv, m, 32);
        int   oc = __shfl_xor(bestc, m, 32);
        if (ov > bestv || (ov == bestv && oc < bestc)) { bestv = ov; bestc = oc; }
      }
      if (l16 == 0) {
        int row = mbase + msub * 16 + r + half * 8;
        unsigned long long key =
            ((unsigned long long)fkey(bestv) << 32) | (unsigned int)(0xFFFFFFFFu - (unsigned)bestc);
        atomicMax(&amax[row], key);
      }
    }
}

__global__ void k_emit(const unsigned long long* __restrict__ amax, int* __restrict__ preds, int t) {
  int b = blockIdx.x * blockDim.x + threadIdx.x;
  if (b < BATCH) {
    unsigned int c = 0xFFFFFFFFu - (unsigned int)(amax[b] & 0xFFFFFFFFull);
    preds[b * TLEN + t] = (int)c;
  }
}

// ---------------------------------------------------------------------------
extern "C" void kernel_launch(void* const* d_in, const int* in_sizes, int n_in,
                              void* d_out, int out_size, void* d_ws, size_t ws_size,
                              hipStream_t stream) {
  (void)in_sizes; (void)n_in; (void)out_size; (void)ws_size;
  const int*   caption = (const int*)  d_in[0];
  const float* img     = (const float*)d_in[1];
  const float* emb     = (const float*)d_in[2];
  const float* W_cond  = (const float*)d_in[3];
  const float* b_cond  = (const float*)d_in[4];
  const float* w_ih    = (const float*)d_in[5];
  const float* w_hh    = (const float*)d_in[6];
  const float* b_ih    = (const float*)d_in[7];
  const float* b_hh    = (const float*)d_in[8];
  const float* W_pred  = (const float*)d_in[9];
  const float* b_pred  = (const float*)d_in[10];
  int* preds = (int*)d_out;

  char* p = (char*)d_ws;
  auto carve = [&](size_t bytes) -> char* {
    char* r = p; p += (bytes + 255) & ~(size_t)255; return r;
  };
  bf16_t* wpb    = (bf16_t*)carve((size_t)VOCAB * NH * 2);
  bf16_t* wihb   = (bf16_t*)carve((size_t)G3 * NH * 2);
  bf16_t* whhb   = (bf16_t*)carve((size_t)G3 * NH * 2);
  bf16_t* wcondb = (bf16_t*)carve((size_t)NH * NC * 2);
  bf16_t* imgb   = (bf16_t*)carve((size_t)BATCH * NC * 2);
  bf16_t* hb     = (bf16_t*)carve((size_t)BATCH * NH * 2);
  bf16_t* xb     = (bf16_t*)carve((size_t)BATCH * NH * 2);
  float*  hf     = (float*) carve((size_t)BATCH * NH * 4);
  float*  gi     = (float*) carve((size_t)BATCH * G3 * 4);
  float*  gh     = (float*) carve((size_t)BATCH * G3 * 4);
  unsigned long long* amax = (unsigned long long*)carve((size_t)BATCH * 8);

  auto cvt = [&](const float* s, bf16_t* d, int n) {
    k_cvt_bf16<<<(n + 255) / 256, 256, 0, stream>>>(s, d, n);
  };
  cvt(W_pred, wpb,    VOCAB * NH);
  cvt(w_ih,   wihb,   G3 * NH);
  cvt(w_hh,   whhb,   G3 * NH);
  cvt(W_cond, wcondb, NH * NC);
  cvt(img,    imgb,   BATCH * NC);

  k_seed<<<1, 256, 0, stream>>>(caption, preds);
  k_h0<<<NH / 64, 128, 0, stream>>>(imgb, wcondb, b_cond, hf, hb);

  const int nblk_logits = (VOCAB + 127) / 128;   // 251
  for (int t = 1; t < TLEN; t++) {
    k_gather<<<(BATCH * NH) / 256, 256, 0, stream>>>(preds, t, emb, xb);
    k_gates<<<dim3(G3 / 64, 2), 128, 0, stream>>>(xb, hb, wihb, whhb, b_ih, b_hh, gi, gh);
    k_combine<<<(BATCH * NH) / 256, 256, 0, stream>>>(gi, gh, hf, hb, amax);
    k_logits<<<nblk_logits, 256, 0, stream>>>(hb, wpb, b_pred, amax);
    k_emit<<<1, 256, 0, stream>>>(amax, preds, t);
  }
}
